// RNNDecoder_377957122338
// MI455X (gfx1250) — matrix operand order repaired and checked by reading, not compiled, not run on hardware
//
#include <hip/hip_runtime.h>
#include <hip/hip_bf16.h>
#include <math.h>

typedef __attribute__((ext_vector_type(8)))  float  v8f;
typedef __attribute__((ext_vector_type(16))) __bf16 v16bf;
typedef __bf16 bf16;

#define N_   32
#define HID_ 512
#define FEAT_ 512
#define CPOS_ 512          // 16*32 spatial positions
#define EMB_ 80
#define VOC_ 554
#define STEPS_ 150
#define G4_  2048          // 4*HID

// ---------------------------------------------------------------------------
// init: zero state, convert W_hh -> bf16 (row-major [j][k])
// ---------------------------------------------------------------------------
__global__ void k_init(const float* __restrict__ W_hh, bf16* __restrict__ whh_bf,
                       float* __restrict__ h, float* __restrict__ c,
                       bf16* __restrict__ h_bf, int* __restrict__ tok) {
    int i = blockIdx.x * blockDim.x + threadIdx.x;
    if (i < G4_ * HID_) whh_bf[i] = (bf16)W_hh[i];
    if (i < N_ * HID_) { h[i] = 0.f; c[i] = 0.f; h_bf[i] = (bf16)0.f; }
    if (i < N_) tok[i] = 0;
}

// ---------------------------------------------------------------------------
// enc_proj[n,c,h] = sum_f rnn_enc[n,f,c] * W_sm[h, 512+f]   (WMMA bf16)
// one wave per 16x16 output tile; 32 n * 32 ctile * 32 htile = 32768 waves
// ---------------------------------------------------------------------------
__global__ void k_encproj(const float* __restrict__ rnn_enc,
                          const float* __restrict__ W_sm,
                          float* __restrict__ enc_proj) {
    int wid  = blockIdx.x * (blockDim.x >> 5) + (threadIdx.x >> 5);
    int lane = threadIdx.x & 31;
    int n  = wid >> 10;
    int rm = wid & 1023;
    int ct = rm >> 5;          // c tile
    int ht = rm & 31;          // h tile
    int half = lane >> 4, lr = lane & 15;

    const float* encA = rnn_enc + (size_t)n * (FEAT_ * CPOS_);   // [f][c]
    v8f acc = {};
    for (int kb = 0; kb < FEAT_; kb += 32) {
        v16bf a, b;
        // A[m][k] = enc[n, c=ct*16+lr, f=kb+k] = rnn_enc[n, kb+k, ct*16+lr]
        #pragma unroll
        for (int e = 0; e < 16; ++e) {
            int k = half * 16 + e;
            a[e] = (bf16)encA[(size_t)(kb + k) * CPOS_ + (ct * 16 + lr)];
        }
        // B[k][nn] = W_sm[(ht*16+nn), 512 + kb + k]
        const float* bp = W_sm + (size_t)(ht * 16 + lr) * (HID_ + FEAT_) + HID_ + kb + half * 16;
        #pragma unroll
        for (int e = 0; e < 16; ++e) b[e] = (bf16)bp[e];
        acc = __builtin_amdgcn_wmma_f32_16x16x32_bf16(false, a, false, b,
                                                      (short)0, acc, false, false);
    }
    // D layout: VGPR r -> M = half*8 + r, N = lr
    float* dp = enc_proj + ((size_t)n * CPOS_ + ct * 16) * HID_ + ht * 16 + lr;
    #pragma unroll
    for (int r = 0; r < 8; ++r)
        dp[(size_t)(half * 8 + r) * HID_] = acc[r];
}

// ---------------------------------------------------------------------------
// gates[n,j] = (h @ W_hh^T)[n,j]  (WMMA bf16, K=512)
//            + emb[tok[n]] . W_ih[j,:] + b_ih[j] + b_hh[j]   (VALU epilogue)
// tiles: mt in {0,1} (n), nt in 0..127 (j) -> 256 waves
// ---------------------------------------------------------------------------
__global__ void k_gates(const bf16* __restrict__ h_bf, const bf16* __restrict__ whh_bf,
                        const float* __restrict__ emb, const int* __restrict__ tok,
                        const float* __restrict__ W_ih, const float* __restrict__ b_ih,
                        const float* __restrict__ b_hh, float* __restrict__ gates) {
    int wid  = blockIdx.x * (blockDim.x >> 5) + (threadIdx.x >> 5);
    int lane = threadIdx.x & 31;
    int mt = wid >> 7;         // n tile (0..1)
    int nt = wid & 127;        // j tile
    int half = lane >> 4, lr = lane & 15;

    v8f acc = {};
    for (int kb = 0; kb < HID_; kb += 32) {
        v16bf a, b;
        const bf16* ap = h_bf + (size_t)(mt * 16 + lr) * HID_ + kb;
        #pragma unroll
        for (int e = 0; e < 16; ++e) {
            int k = half * 16 + e;
            a[e] = ap[k];
        }
        const bf16* bp = whh_bf + (size_t)(nt * 16 + lr) * HID_ + kb + half * 16;
        #pragma unroll
        for (int e = 0; e < 16; ++e) b[e] = bp[e];
        acc = __builtin_amdgcn_wmma_f32_16x16x32_bf16(false, a, false, b,
                                                      (short)0, acc, false, false);
    }
    int j = nt * 16 + lr;
    float bias = b_ih[j] + b_hh[j];
    const float* wrow = W_ih + (size_t)j * EMB_;
    #pragma unroll 1
    for (int r = 0; r < 8; ++r) {
        int m = mt * 16 + half * 8 + r;                 // batch index
        const float* x = emb + (size_t)tok[m] * EMB_;
        float s = acc[r] + bias;
        for (int e = 0; e < EMB_; ++e) s += x[e] * wrow[e];
        gates[(size_t)m * G4_ + j] = s;
    }
}

__device__ __forceinline__ float sigm(float x) { return 1.f / (1.f + __expf(-x)); }

// ---------------------------------------------------------------------------
// LSTM cell update; also refresh bf16 copy of h
// ---------------------------------------------------------------------------
__global__ void k_cell(const float* __restrict__ gates, float* __restrict__ c,
                       float* __restrict__ h, bf16* __restrict__ h_bf) {
    int i = blockIdx.x * blockDim.x + threadIdx.x;     // 0..16383
    int n = i >> 9, k = i & (HID_ - 1);
    const float* g = gates + (size_t)n * G4_;
    float gi = g[k], gf = g[HID_ + k], gg = g[2 * HID_ + k], go = g[3 * HID_ + k];
    float cn = sigm(gf) * c[i] + sigm(gi) * tanhf(gg);
    float hn = sigm(go) * tanhf(cn);
    c[i] = cn; h[i] = hn; h_bf[i] = (bf16)hn;
}

// ---------------------------------------------------------------------------
// q[n,k] = b_sm[k] + h[n,:] . W_sm[k, :512]
// ---------------------------------------------------------------------------
__global__ void k_q(const float* __restrict__ h, const float* __restrict__ W_sm,
                    const float* __restrict__ b_sm, float* __restrict__ q) {
    int i = blockIdx.x * blockDim.x + threadIdx.x;     // 0..16383
    int n = i >> 9, k = i & (HID_ - 1);
    const float* hp = h + (size_t)n * HID_;
    const float* wp = W_sm + (size_t)k * (HID_ + FEAT_);
    float s = b_sm[k];
    for (int kk = 0; kk < HID_; ++kk) s += hp[kk] * wp[kk];
    q[i] = s;
}

// ---------------------------------------------------------------------------
// score[n,c] = b_sv + sum_h w_sv[h] * tanh(enc_proj[n,c,h] + q[n,h])
// one wave per (n,c); enc_proj stays L2-resident (32MB < 192MB L2)
// ---------------------------------------------------------------------------
__global__ void k_score(const float* __restrict__ enc_proj, const float* __restrict__ q,
                        const float* __restrict__ w_sv, const float* __restrict__ b_sv,
                        float* __restrict__ score) {
    int wid  = blockIdx.x * (blockDim.x >> 5) + (threadIdx.x >> 5);
    int lane = threadIdx.x & 31;
    int n = wid >> 9, cpos = wid & (CPOS_ - 1);
    const float* ep = enc_proj + ((size_t)n * CPOS_ + cpos) * HID_;
    const float* qp = q + (size_t)n * HID_;
    float s = 0.f;
    #pragma unroll 4
    for (int t = 0; t < HID_ / 32; ++t) {
        int hh = lane + 32 * t;
        s += w_sv[hh] * tanhf(ep[hh] + qp[hh]);
    }
    #pragma unroll
    for (int m = 16; m >= 1; m >>= 1) s += __shfl_xor(s, m, 32);
    if (lane == 0) score[(size_t)n * CPOS_ + cpos] = s + b_sv[0];
}

// ---------------------------------------------------------------------------
// softmax over C=512 (in place), one block of 256 per n
// ---------------------------------------------------------------------------
__global__ void k_softmax(float* __restrict__ score) {
    __shared__ float red[256];
    int n = blockIdx.x, tid = threadIdx.x;
    float* sp = score + (size_t)n * CPOS_;
    float v0 = sp[tid], v1 = sp[tid + 256];
    red[tid] = fmaxf(v0, v1); __syncthreads();
    for (int s = 128; s >= 1; s >>= 1) {
        if (tid < s) red[tid] = fmaxf(red[tid], red[tid + s]);
        __syncthreads();
    }
    float mx = red[0]; __syncthreads();
    float e0 = __expf(v0 - mx), e1 = __expf(v1 - mx);
    red[tid] = e0 + e1; __syncthreads();
    for (int s = 128; s >= 1; s >>= 1) {
        if (tid < s) red[tid] += red[tid + s];
        __syncthreads();
    }
    float inv = 1.f / red[0];
    sp[tid] = e0 * inv; sp[tid + 256] = e1 * inv;
}

// ---------------------------------------------------------------------------
// ctx[n,f] = sum_c attn[n,c] * rnn_enc[n,f,c]   (c contiguous!)
// ---------------------------------------------------------------------------
__global__ void k_ctx(const float* __restrict__ attn, const float* __restrict__ rnn_enc,
                      float* __restrict__ ctx) {
    int wid  = blockIdx.x * (blockDim.x >> 5) + (threadIdx.x >> 5);
    int lane = threadIdx.x & 31;
    int n = wid >> 9, f = wid & (FEAT_ - 1);
    const float* ap = attn + (size_t)n * CPOS_;
    const float* ep = rnn_enc + ((size_t)n * FEAT_ + f) * CPOS_;
    float s = 0.f;
    #pragma unroll 4
    for (int t = 0; t < CPOS_ / 32; ++t) {
        int cc = lane + 32 * t;
        s += ap[cc] * ep[cc];
    }
    #pragma unroll
    for (int m = 16; m >= 1; m >>= 1) s += __shfl_xor(s, m, 32);
    if (lane == 0) ctx[(size_t)n * FEAT_ + f] = s;
}

// ---------------------------------------------------------------------------
// out = tanh([h,ctx] @ W_ctx^T + b_ctx); logits = out @ W_out^T + b_out;
// log_softmax -> d_out[n, v, t]; argmax -> tok[n]. One block of 256 per n.
// ---------------------------------------------------------------------------
__global__ void k_out(const float* __restrict__ h, const float* __restrict__ ctx,
                      const float* __restrict__ W_ctx, const float* __restrict__ b_ctx,
                      const float* __restrict__ W_out, const float* __restrict__ b_out,
                      float* __restrict__ out, int* __restrict__ tok, int t) {
    __shared__ float outv[EMB_];
    __shared__ float logits[VOC_];
    __shared__ float rmax[256];
    __shared__ int   ridx[256];
    __shared__ float rsum[256];
    int n = blockIdx.x, tid = threadIdx.x;

    if (tid < EMB_) {
        const float* w  = W_ctx + (size_t)tid * (HID_ + FEAT_);
        const float* hp = h + (size_t)n * HID_;
        const float* cp = ctx + (size_t)n * FEAT_;
        float s = b_ctx[tid];
        for (int k = 0; k < HID_;  ++k) s += hp[k] * w[k];
        for (int k = 0; k < FEAT_; ++k) s += cp[k] * w[HID_ + k];
        outv[tid] = tanhf(s);
    }
    __syncthreads();

    float lmax = -INFINITY; int lidx = 0x7fffffff;
    for (int v = tid; v < VOC_; v += 256) {
        const float* w = W_out + (size_t)v * EMB_;
        float s = b_out[v];
        for (int e = 0; e < EMB_; ++e) s += outv[e] * w[e];
        logits[v] = s;
        if (s > lmax) { lmax = s; lidx = v; }   // strided v increasing -> first-max kept
    }
    rmax[tid] = lmax; ridx[tid] = lidx; __syncthreads();
    for (int s = 128; s >= 1; s >>= 1) {
        if (tid < s) {
            float a = rmax[tid], b = rmax[tid + s];
            if (b > a || (b == a && ridx[tid + s] < ridx[tid])) {
                rmax[tid] = b; ridx[tid] = ridx[tid + s];
            }
        }
        __syncthreads();
    }
    float mx = rmax[0]; int amax = ridx[0]; __syncthreads();
    float ls = 0.f;
    for (int v = tid; v < VOC_; v += 256) ls += __expf(logits[v] - mx);
    rsum[tid] = ls; __syncthreads();
    for (int s = 128; s >= 1; s >>= 1) {
        if (tid < s) rsum[tid] += rsum[tid + s];
        __syncthreads();
    }
    float lse = mx + __logf(rsum[0]);
    for (int v = tid; v < VOC_; v += 256)
        out[((size_t)n * VOC_ + v) * STEPS_ + t] = logits[v] - lse;
    if (tid == 0) tok[n] = amax;
}

// ---------------------------------------------------------------------------
extern "C" void kernel_launch(void* const* d_in, const int* in_sizes, int n_in,
                              void* d_out, int out_size, void* d_ws, size_t ws_size,
                              hipStream_t stream) {
    const float* rnn_enc = (const float*)d_in[0];
    const float* emb     = (const float*)d_in[1];
    const float* W_ih    = (const float*)d_in[2];
    const float* W_hh    = (const float*)d_in[3];
    const float* b_ih    = (const float*)d_in[4];
    const float* b_hh    = (const float*)d_in[5];
    const float* W_sm    = (const float*)d_in[6];
    const float* b_sm    = (const float*)d_in[7];
    const float* w_sv    = (const float*)d_in[8];
    const float* b_sv    = (const float*)d_in[9];
    const float* W_ctx   = (const float*)d_in[10];
    const float* b_ctx   = (const float*)d_in[11];
    const float* W_out   = (const float*)d_in[12];
    const float* b_out   = (const float*)d_in[13];
    float* out = (float*)d_out;

    char* ws = (char*)d_ws;
    float* enc_proj = (float*)(ws + 0);                      // 33,554,432 B
    float* gates    = (float*)(ws + 33554432);               //    262,144 B
    float* h        = (float*)(ws + 33816576);               //     65,536 B
    float* c        = (float*)(ws + 33882112);               //     65,536 B
    float* q        = (float*)(ws + 33947648);               //     65,536 B
    float* attn     = (float*)(ws + 34013184);               //     65,536 B (score->attn)
    float* ctx      = (float*)(ws + 34078720);               //     65,536 B
    bf16*  h_bf     = (bf16*) (ws + 34144256);               //     32,768 B
    bf16*  whh_bf   = (bf16*) (ws + 34177024);               //  2,097,152 B
    int*   tok      = (int*)  (ws + 36274176);               //        128 B

    // prologue: init state + W_hh->bf16, then WMMA enc projection
    k_init<<<4096, 256, 0, stream>>>(W_hh, whh_bf, h, c, h_bf, tok);
    k_encproj<<<4096, 256, 0, stream>>>(rnn_enc, W_sm, enc_proj);

    for (int t = 0; t < STEPS_; ++t) {
        k_gates  <<<  32, 256, 0, stream>>>(h_bf, whh_bf, emb, tok, W_ih, b_ih, b_hh, gates);
        k_cell   <<<  64, 256, 0, stream>>>(gates, c, h, h_bf);
        k_q      <<<  64, 256, 0, stream>>>(h, W_sm, b_sm, q);
        k_score  <<<2048, 256, 0, stream>>>(enc_proj, q, w_sv, b_sv, attn);
        k_softmax<<<  32, 256, 0, stream>>>(attn);
        k_ctx    <<<2048, 256, 0, stream>>>(attn, rnn_enc, ctx);
        k_out    <<<  32, 256, 0, stream>>>(h, ctx, W_ctx, b_ctx, W_out, b_out, out, tok, t);
    }
}
